// OPTAttentionSeq_9981503995995
// MI455X (gfx1250) — compile-verified, matrix operand order checked by reading, not run on hardware
//
#include <hip/hip_runtime.h>
#include <hip/hip_bf16.h>

typedef __bf16 bf16;
typedef __attribute__((ext_vector_type(16))) __bf16 v16bf;
typedef __attribute__((ext_vector_type(8)))  __bf16 v8bf;
typedef __attribute__((ext_vector_type(8)))  float  v8f;

union Frag { v16bf v; v8bf h[2]; };

static constexpr int Bn = 2;     // batch
static constexpr int T  = 2048;  // seq len
static constexpr int E  = 2048;  // embed
static constexpr int H  = 32;    // heads
static constexpr int D  = 64;    // head dim

__device__ __forceinline__ v8f wmma_bf16(v16bf a, v16bf b, v8f c) {
  // 8 args: (neg_a, A, neg_b, B, c_mod, C, reuse_a, reuse_b)
  return __builtin_amdgcn_wmma_f32_16x16x32_bf16(false, a, false, b, (short)0, c,
                                                 false, false);
}

// Async copy 16B global -> LDS, tracked by ASYNCcnt (CDNA5).
__device__ __forceinline__ void async_b128(const bf16* gptr, unsigned lds_byte_addr) {
  asm volatile("global_load_async_to_lds_b128 %0, %1, off"
               :: "v"(lds_byte_addr), "v"(gptr)
               : "memory");
}
#define WAIT_ASYNC(n) asm volatile("s_wait_asynccnt %0" :: "i"(n) : "memory")

__device__ __forceinline__ unsigned lds_addr(const void* p) {
  // generic LDS pointer: low 32 bits are the hardware LDS byte address
  return (unsigned)(uintptr_t)p;
}

__device__ __forceinline__ float hmax16(float v) {
#pragma unroll
  for (int m = 1; m < 16; m <<= 1) v = fmaxf(v, __shfl_xor(v, m, 32));
  return v;
}
__device__ __forceinline__ float hsum16(float v) {
#pragma unroll
  for (int m = 1; m < 16; m <<= 1) v += __shfl_xor(v, m, 32);
  return v;
}

// ---------------------------------------------------------------- convert ---
__global__ __launch_bounds__(256) void cvt_f32_bf16(const float* __restrict__ in,
                                                    bf16* __restrict__ out,
                                                    size_t n) {
  size_t i = (size_t)blockIdx.x * blockDim.x + threadIdx.x;
  size_t stride = (size_t)gridDim.x * blockDim.x;
  for (; i < n; i += stride) out[i] = (bf16)in[i];
}

// ------------------------------------------------------------------- GEMM ---
// C[M=4096, N=2048] = X[M,E] * W[N,E]^T + bias
// W tile (64 rows x 32 k, 4KB) staged in double-buffered LDS by async DMA,
// shared by all 8 waves of the block (8x L2 traffic reduction).
// mode 0: q (scale 1/8, split-head bf16)   mode 1: k (split-head bf16)
// mode 2: v (per-head transposed bf16)     mode 3: fp32 output projection
__global__ __launch_bounds__(256) void gemm_bf16(const bf16* __restrict__ X,
                                                 const bf16* __restrict__ W,
                                                 const float* __restrict__ bias,
                                                 bf16* __restrict__ outb,
                                                 float* __restrict__ outf,
                                                 int mode) {
  __shared__ alignas(16) bf16 wtile[2][64 * 32];

  const int tid    = threadIdx.x;
  const int lane   = tid & 31;
  const int wave   = tid >> 5;
  const int lane15 = lane & 15;
  const int hi     = lane >> 4;
  const int koff   = hi ? 8 : 0;
  const int mBase  = (blockIdx.y * 8 + wave) * 16;
  const int nBase  = blockIdx.x * 64;

  // staging assignment: 256 threads x 16B = one 4KB tile
  const int srow   = tid >> 2;          // 0..63 : W row within tile
  const int scol   = (tid & 3) * 8;     // bf16 column within 32-wide k-chunk
  const bf16* wsrc = W + (size_t)(nBase + srow) * E + scol;
  const unsigned soff = (unsigned)((srow * 32 + scol) * sizeof(bf16));
  const unsigned wbuf0 = lds_addr(&wtile[0][0]) + soff;
  const unsigned wbuf1 = lds_addr(&wtile[1][0]) + soff;

  v8f acc[4] = {};
  const bf16* xrow = X + (size_t)(mBase + lane15) * E;

  auto compute_tile = [&](int kt) {
    const bf16* wt = &wtile[kt & 1][0];
    Frag a;
    a.h[0] = *(const v8bf*)(xrow + kt * 32 + koff);
    a.h[1] = *(const v8bf*)(xrow + kt * 32 + koff + 16);
    Frag bfr[4];
#pragma unroll
    for (int j = 0; j < 4; ++j) {
      const bf16* wr = wt + (j * 16 + lane15) * 32;
      bfr[j].h[0] = *(const v8bf*)(wr + koff);
      bfr[j].h[1] = *(const v8bf*)(wr + koff + 16);
    }
#pragma unroll
    for (int j = 0; j < 4; ++j)
      acc[j] = wmma_bf16(a.v, bfr[j].v, acc[j]);
  };

  constexpr int KT = E / 32;  // 64 k-steps
  async_b128(wsrc, wbuf0);    // prologue: tile 0 -> buf 0

  // steady state: branch-free body (last iteration peeled)
  for (int kt = 0; kt < KT - 1; ++kt) {
    async_b128(wsrc + (kt + 1) * 32, ((kt + 1) & 1) ? wbuf1 : wbuf0);
    WAIT_ASYNC(1);            // tile kt has landed (in-order completion)
    __syncthreads();
    compute_tile(kt);
    __syncthreads();          // buffer safe to overwrite next iteration
  }
  WAIT_ASYNC(0);
  __syncthreads();
  compute_tile(KT - 1);

#pragma unroll
  for (int j = 0; j < 4; ++j) {
    const int e  = nBase + j * 16 + lane15;
    const float bv = bias[e];
#pragma unroll
    for (int r = 0; r < 8; ++r) {
      const int g  = mBase + r + hi * 8;   // global row in [0, B*T)
      const int bb = g >> 11;              // T == 2048
      const int t  = g & (T - 1);
      float val = acc[j][r] + bv;
      if (mode == 0) {
        val *= 0.125f;  // D^-0.5, D = 64
        const int h = e >> 6, d = e & 63;
        outb[(((size_t)(bb * H + h) * T) + t) * D + d] = (bf16)val;
      } else if (mode == 1) {
        const int h = e >> 6, d = e & 63;
        outb[(((size_t)(bb * H + h) * T) + t) * D + d] = (bf16)val;
      } else if (mode == 2) {
        const int h = e >> 6, d = e & 63;
        outb[(((size_t)(bb * H + h) * D) + d) * T + t] = (bf16)val;
      } else {
        outf[(size_t)g * E + e] = val;
      }
    }
  }
}

// -------------------------------------------------------- flash attention ---
// 4 waves per block share one (b,h): K/V chunks async-staged in LDS once per
// block (4x traffic reduction), double-buffered over the 64 s-chunks.
__global__ __launch_bounds__(128) void flash_attn(const bf16* __restrict__ Q,
                                                  const bf16* __restrict__ K,
                                                  const bf16* __restrict__ VT,
                                                  const float* __restrict__ mask,
                                                  bf16* __restrict__ O) {
  __shared__ alignas(16) bf16 ktile[2][32 * 64];  // [s][d]
  __shared__ alignas(16) bf16 vtile[2][64 * 32];  // [d][s]
  __shared__ alignas(16) bf16 plds[4][16 * 32];

  const int tid    = threadIdx.x;
  const int lane   = tid & 31;
  const int wave   = tid >> 5;
  const int lane15 = lane & 15;
  const int hi     = lane >> 4;
  const int koff   = hi ? 8 : 0;
  const int bh     = blockIdx.y;          // b*H + h
  const int b      = bh >> 5;
  const int h      = bh & 31;
  const int tBase  = (blockIdx.x * 4 + wave) * 16;
  bf16* myP = plds[wave];

  // staging assignment (128 threads): K chunk 32x64, V chunk 64x32, 4KB each
  const int krow = tid >> 2;              // 0..31
  const int kcol = (tid & 3) * 16;        // 2 chunks: kcol, kcol+8
  const bf16* ksrc = K + ((size_t)bh * T + krow) * D + kcol;
  const unsigned kof = (unsigned)((krow * 64 + kcol) * sizeof(bf16));

  const int vrow = tid >> 1;              // 0..63
  const int vcol = (tid & 1) * 16;        // 2 chunks: vcol, vcol+8
  const bf16* vsrc = VT + ((size_t)bh * D + vrow) * T + vcol;
  const unsigned vof = (unsigned)((vrow * 32 + vcol) * sizeof(bf16));

  const unsigned kbuf[2] = { lds_addr(&ktile[0][0]) + kof, lds_addr(&ktile[1][0]) + kof };
  const unsigned vbuf[2] = { lds_addr(&vtile[0][0]) + vof, lds_addr(&vtile[1][0]) + vof };

  // Q fragments: 16 rows x 64 K, as two 16x32 A-fragments
  Frag aq[2];
  {
    const bf16* qp = Q + ((size_t)bh * T + (tBase + lane15)) * D;
#pragma unroll
    for (int c = 0; c < 2; ++c) {
      aq[c].h[0] = *(const v8bf*)(qp + c * 32 + koff);
      aq[c].h[1] = *(const v8bf*)(qp + c * 32 + koff + 16);
    }
  }

  v8f o[4] = {};
  float m[8], l[8];
#pragma unroll
  for (int r = 0; r < 8; ++r) { m[r] = -__builtin_huge_valf(); l[r] = 0.0f; }

  const float* mrow = mask + (size_t)b * T * T;

  auto stage_chunk = [&](int st) {
    const int nb = st & 1;
    const size_t kadv = (size_t)st * 32 * D;  // 32 key rows forward
    const size_t vadv = (size_t)st * 32;      // 32 value cols forward
    async_b128(ksrc + kadv,     kbuf[nb]);
    async_b128(ksrc + kadv + 8, kbuf[nb] + 16);
    async_b128(vsrc + vadv,     vbuf[nb]);
    async_b128(vsrc + vadv + 8, vbuf[nb] + 16);
  };

  auto attn_step = [&](int st) {
    const int s0 = st * 32;
    const bf16* kt = &ktile[st & 1][0];
    const bf16* vt = &vtile[st & 1][0];

    // ---- scores: two 16x16 tiles over K = D = 64 (batch loads, then WMMAs)
    Frag kf[2][2];
#pragma unroll
    for (int j = 0; j < 2; ++j) {
      const bf16* kp = kt + (j * 16 + lane15) * 64;
      kf[j][0].h[0] = *(const v8bf*)(kp + koff);
      kf[j][0].h[1] = *(const v8bf*)(kp + koff + 16);
      kf[j][1].h[0] = *(const v8bf*)(kp + 32 + koff);
      kf[j][1].h[1] = *(const v8bf*)(kp + 32 + koff + 16);
    }
    v8f sc[2];
#pragma unroll
    for (int j = 0; j < 2; ++j) {
      v8f z = {};
      z = wmma_bf16(aq[0].v, kf[j][0].v, z);
      sc[j] = wmma_bf16(aq[1].v, kf[j][1].v, z);
    }

    // ---- mask add + clamp
#pragma unroll
    for (int j = 0; j < 2; ++j) {
#pragma unroll
      for (int r = 0; r < 8; ++r) {
        const int t = tBase + r + hi * 8;
        const int s = s0 + j * 16 + lane15;
        float v = sc[j][r] + mrow[(size_t)t * T + s];
        sc[j][r] = fmaxf(v, -3.402823466e+38f);
      }
    }

    // ---- online softmax update
#pragma unroll
    for (int r = 0; r < 8; ++r) {
      const float x0 = sc[0][r], x1 = sc[1][r];
      const float mx   = hmax16(fmaxf(x0, x1));
      const float mnew = fmaxf(m[r], mx);
      const float corr = __expf(m[r] - mnew);
      const float p0 = __expf(x0 - mnew);
      const float p1 = __expf(x1 - mnew);
      const float rs = hsum16(p0 + p1);
      l[r] = l[r] * corr + rs;
      m[r] = mnew;
#pragma unroll
      for (int f = 0; f < 4; ++f) o[f][r] *= corr;
      const int row = r + hi * 8;
      myP[row * 32 + lane15]      = (bf16)p0;
      myP[row * 32 + 16 + lane15] = (bf16)p1;
    }

    asm volatile("s_wait_dscnt 0" ::: "memory");

    // ---- re-read P as a 16x32 A-fragment
    Frag pf;
    pf.h[0] = *(const v8bf*)(myP + lane15 * 32 + koff);
    pf.h[1] = *(const v8bf*)(myP + lane15 * 32 + koff + 16);

    // ---- accumulate P @ V  (V chunk in LDS, [d][s]; batch loads, then WMMAs)
    Frag vf[4];
#pragma unroll
    for (int j = 0; j < 4; ++j) {
      const bf16* vp = vt + (j * 16 + lane15) * 32;
      vf[j].h[0] = *(const v8bf*)(vp + koff);
      vf[j].h[1] = *(const v8bf*)(vp + koff + 16);
    }
#pragma unroll
    for (int j = 0; j < 4; ++j)
      o[j] = wmma_bf16(pf.v, vf[j].v, o[j]);
  };

  constexpr int NS = T / 32;  // 64 s-chunks
  stage_chunk(0);             // prologue

  // steady state: branch-free body (last chunk peeled)
  for (int st = 0; st < NS - 1; ++st) {
    stage_chunk(st + 1);
    WAIT_ASYNC(4);            // chunk st has landed
    __syncthreads();
    attn_step(st);
    __syncthreads();          // K/V buffer reusable next iteration
  }
  WAIT_ASYNC(0);
  __syncthreads();
  attn_step(NS - 1);

  // ---- epilogue: divide by l, merge heads: O[b, t, h*64 + d]
#pragma unroll
  for (int r = 0; r < 8; ++r) {
    const float rl = 1.0f / l[r];
    const int t = tBase + r + hi * 8;
#pragma unroll
    for (int j = 0; j < 4; ++j) {
      const int e = h * D + j * 16 + lane15;
      O[((size_t)(b * T + t)) * E + e] = (bf16)(o[j][r] * rl);
    }
  }
}

// ------------------------------------------------------------------ launch ---
extern "C" void kernel_launch(void* const* d_in, const int* in_sizes, int n_in,
                              void* d_out, int out_size, void* d_ws, size_t ws_size,
                              hipStream_t stream) {
  const float* hs   = (const float*)d_in[0];
  const float* mask = (const float*)d_in[1];
  const float* Wq   = (const float*)d_in[2];
  const float* bq   = (const float*)d_in[3];
  const float* Wk   = (const float*)d_in[4];
  const float* bk   = (const float*)d_in[5];
  const float* Wv   = (const float*)d_in[6];
  const float* bv   = (const float*)d_in[7];
  const float* Wo   = (const float*)d_in[8];
  const float* bo   = (const float*)d_in[9];
  float* out = (float*)d_out;

  const size_t nHid = (size_t)Bn * T * E;   // 8Mi elements
  const size_t nW   = (size_t)E * E;        // 4Mi elements

  char* ws = (char*)d_ws;
  size_t off = 0;
  bf16* xb  = (bf16*)(ws + off); off += nHid * sizeof(bf16);
  bf16* wqb = (bf16*)(ws + off); off += nW * sizeof(bf16);
  bf16* wkb = (bf16*)(ws + off); off += nW * sizeof(bf16);
  bf16* wvb = (bf16*)(ws + off); off += nW * sizeof(bf16);
  bf16* wob = (bf16*)(ws + off); off += nW * sizeof(bf16);
  bf16* qb  = (bf16*)(ws + off); off += nHid * sizeof(bf16);
  bf16* kb  = (bf16*)(ws + off); off += nHid * sizeof(bf16);
  bf16* vtb = (bf16*)(ws + off); off += nHid * sizeof(bf16);
  bf16* ab  = (bf16*)(ws + off); off += nHid * sizeof(bf16);

  // 1) convert inputs to bf16
  cvt_f32_bf16<<<4096, 256, 0, stream>>>(hs, xb, nHid);
  cvt_f32_bf16<<<2048, 256, 0, stream>>>(Wq, wqb, nW);
  cvt_f32_bf16<<<2048, 256, 0, stream>>>(Wk, wkb, nW);
  cvt_f32_bf16<<<2048, 256, 0, stream>>>(Wv, wvb, nW);
  cvt_f32_bf16<<<2048, 256, 0, stream>>>(Wo, wob, nW);

  // 2) q/k/v projections (grid: N/64 x M/(16*8))
  dim3 ggrid(E / 64, (Bn * T) / (16 * 8));
  gemm_bf16<<<ggrid, 256, 0, stream>>>(xb, wqb, bq, qb, nullptr, 0);
  gemm_bf16<<<ggrid, 256, 0, stream>>>(xb, wkb, bk, kb, nullptr, 1);
  gemm_bf16<<<ggrid, 256, 0, stream>>>(xb, wvb, bv, vtb, nullptr, 2);

  // 3) flash attention: grid (T/16/4, B*H), 4 waves per block
  flash_attn<<<dim3(T / 16 / 4, Bn * H), 128, 0, stream>>>(qb, kb, vtb, mask, ab);

  // 4) output projection -> fp32
  gemm_bf16<<<ggrid, 256, 0, stream>>>(ab, wob, bo, nullptr, out, 3);
}
